// LocalPointDecoder_63788854280441
// MI455X (gfx1250) — matrix operand order repaired and checked by reading, not compiled, not run on hardware
//
#include <hip/hip_runtime.h>

#define BQ   8192
#define BB   2
#define PN   4096
#define CDN  128
#define HD   256
#define NBK  5

#define WCHUNK (256 * 32)   // f16 elems per packed weight K-chunk  ([n][k], stride 32)
#define FCHUNK (128 * 32)   // f16 elems per packed fea chunk       ([cd][p], stride 32)
#define CBUF_ELEMS ((size_t)BB * BQ * CDN)
#define WPACK_OFF  CBUF_ELEMS
#define NWCHUNKS   100      // 20 (fc_c) + 40 (blk0) + 40 (blk1)
#define FPACK_OFF  (WPACK_OFF + (size_t)NWCHUNKS * WCHUNK)

typedef __attribute__((ext_vector_type(16))) _Float16 v16h;
typedef __attribute__((ext_vector_type(8)))  _Float16 v8h;
typedef __attribute__((ext_vector_type(8)))  float    v8f;

// ---- WMMA fragment loaders (CDNA5 16x16x32 f16 layouts, CDNA5 ISA §7.12.2) ----

// A (16x32 f16) from row-major LDS [rows][stride]; two contiguous 8xK runs per lane.
__device__ __forceinline__ v16h frag_a_lds(const _Float16* lds, int row_base, int k_base, int stride) {
  int lane = threadIdx.x & 31;
  int m    = lane & 15;
  int half = lane >> 4;
  const _Float16* rp = lds + (row_base + m) * stride + k_base + half * 8;
  v8h lo = *(const v8h*)(rp);        // K = half*8   .. +7   -> V0..3
  v8h hi = *(const v8h*)(rp + 16);   // K = 16+half*8 .. +7  -> V4..7
  return __builtin_shufflevector(lo, hi, 0, 1, 2, 3, 4, 5, 6, 7, 8, 9, 10, 11, 12, 13, 14, 15);
}

// B (32x16 f16) DIRECT from packed global image [n][32k]; 32 contiguous f16 per lane.
__device__ __forceinline__ v16h frag_b_g(const _Float16* __restrict__ gT, int n_base) {
  int lane = threadIdx.x & 31;
  int n    = lane & 15;
  int half = lane >> 4;
  const _Float16* cp = gT + (n_base + n) * 32 + half * 8;
  v8h lo = *(const v8h*)(cp);
  v8h hi = *(const v8h*)(cp + 16);
  return __builtin_shufflevector(lo, hi, 0, 1, 2, 3, 4, 5, 6, 7, 8, 9, 10, 11, 12, 13, 14, 15);
}

// transpose one 32(k) x NCOLS(n) f32 panel into packed f16 [n][k] stride-32 image
template <int NCOLS>
__device__ __forceinline__ void repack_panel(const float* __restrict__ src, _Float16* __restrict__ dst, int tid) {
  constexpr int NTILES = (NCOLS / 4) * 4;        // 4n x 8k micro-tiles
#pragma unroll
  for (int it = 0; it < NTILES / 128; ++it) {
    int tile = it * 128 + tid;
    int nb = (tile % (NCOLS / 4)) * 4;
    int kb = (tile / (NCOLS / 4)) * 8;
    v8h tmp[4];
#pragma unroll
    for (int kk = 0; kk < 8; ++kk) {
      float4 v = *(const float4*)(src + (size_t)(kb + kk) * NCOLS + nb);
      tmp[0][kk] = (_Float16)v.x; tmp[1][kk] = (_Float16)v.y;
      tmp[2][kk] = (_Float16)v.z; tmp[3][kk] = (_Float16)v.w;
    }
#pragma unroll
    for (int nn = 0; nn < 4; ++nn)
      *(v8h*)(dst + (nb + nn) * 32 + kb) = tmp[nn];
  }
}

// ================= Kernel 0a: one-time weight repack ===========
__global__ __launch_bounds__(128)
void k_repack_w(const float* __restrict__ fc_c_w, const float* __restrict__ blk0_w,
                const float* __restrict__ blk1_w, _Float16* __restrict__ wpack) {
  const int id  = blockIdx.x;       // 0..99
  const int tid = threadIdx.x;
  const float* src;
  if (id < 20)      { int i = id >> 2,        ck = id & 3;        src = fc_c_w + (size_t)i * CDN * HD + (size_t)ck * 32 * HD; }
  else if (id < 60) { int i = (id - 20) >> 3, ck = (id - 20) & 7; src = blk0_w + (size_t)i * HD  * HD + (size_t)ck * 32 * HD; }
  else              { int i = (id - 60) >> 3, ck = (id - 60) & 7; src = blk1_w + (size_t)i * HD  * HD + (size_t)ck * 32 * HD; }
  repack_panel<HD>(src, wpack + (size_t)id * WCHUNK, tid);
}

// ================= Kernel 0b: one-time fea repack =============
__global__ __launch_bounds__(128)
void k_repack_fea(const float* __restrict__ fea, _Float16* __restrict__ fpack) {
  const int id  = blockIdx.x;       // b*128 + pchunk, 0..255
  const int tid = threadIdx.x;
  const int b   = id >> 7;
  const int pc  = (id & 127) * 32;
  repack_panel<CDN>(fea + (size_t)(b * PN + pc) * CDN, fpack + (size_t)id * FCHUNK, tid);
}

// ================= Kernel 1: fused gaussian-weight GEMM  c = (W @ fea) / rowsum(W) =================
// grid (BQ/32, B), 128 threads (4 waves). Wave w: rows 16*(w&1), cols 64*(w>>1).
__global__ __launch_bounds__(128)
void k_gauss_feature(const float* __restrict__ p, const float* __restrict__ pp,
                     const _Float16* __restrict__ fpack, _Float16* __restrict__ cbuf) {
  __shared__ float     s_p[96];
  __shared__ float     s_pp[96];
  __shared__ float     s_wsum[32];
  __shared__ _Float16  s_wA[32 * 40];     // 32q x 32p weight tile, row-major (A)

  const int b    = blockIdx.y;
  const int q0   = blockIdx.x * 32;
  const int tid  = threadIdx.x;
  const int lane = tid & 31;
  const int wave = tid >> 5;

  if (tid < 96) s_p[tid] = p[(size_t)(b * BQ + q0) * 3 + tid];
  if (tid < 32) s_wsum[tid] = 0.0f;

  const v8f vzero = {0.f, 0.f, 0.f, 0.f, 0.f, 0.f, 0.f, 0.f};
  v8f acc[4] = {vzero, vzero, vzero, vzero};

  const int rowA = 16 * (wave & 1);
  const int colB = 64 * (wave >> 1);

  for (int pc = 0; pc < PN; pc += 32) {
    __syncthreads();                         // previous WMMA reads of s_wA done
    if (tid < 96) s_pp[tid] = pp[(size_t)(b * PN + pc) * 3 + tid];
    __syncthreads();                         // s_pp staged
    {
      int   qi = tid >> 2;
      float qx = s_p[qi*3+0], qy = s_p[qi*3+1], qz = s_p[qi*3+2];
      float partial = 0.f;
#pragma unroll
      for (int j = 0; j < 8; ++j) {
        int   pi = (tid & 3) * 8 + j;
        float dx = s_pp[pi*3+0] - qx;
        float dy = s_pp[pi*3+1] - qy;
        float dz = s_pp[pi*3+2] - qz;
        float d2 = dx*dx + dy*dy + dz*dz;
        float dist = sqrtf(d2) + 1e-5f;
        float w = __expf(-dist * dist * 100.0f);   // 1/VAR = 100
        s_wA[qi * 40 + pi] = (_Float16)w;
        partial += w;
      }
      atomicAdd(&s_wsum[qi], partial);       // ds_add_f32
    }
    __syncthreads();                         // w tile ready
    const _Float16* fc = fpack + (size_t)(b * 128 + (pc >> 5)) * FCHUNK;
    v16h a  = frag_a_lds(s_wA, rowA, 0, 40);
    v16h b0 = frag_b_g(fc, colB +  0);
    v16h b1 = frag_b_g(fc, colB + 16);
    v16h b2 = frag_b_g(fc, colB + 32);
    v16h b3 = frag_b_g(fc, colB + 48);
    acc[0] = __builtin_amdgcn_wmma_f32_16x16x32_f16(false, a, false, b0, (short)0, acc[0], false, false);
    acc[1] = __builtin_amdgcn_wmma_f32_16x16x32_f16(false, a, false, b1, (short)0, acc[1], false, false);
    acc[2] = __builtin_amdgcn_wmma_f32_16x16x32_f16(false, a, false, b2, (short)0, acc[2], false, false);
    acc[3] = __builtin_amdgcn_wmma_f32_16x16x32_f16(false, a, false, b3, (short)0, acc[3], false, false);
  }
  __syncthreads();

  const int n  = lane & 15;
  const int mh = (lane >> 4) * 8;
  float inv[8];
#pragma unroll
  for (int r = 0; r < 8; ++r) inv[r] = __builtin_amdgcn_rcpf(s_wsum[rowA + mh + r]);
#pragma unroll
  for (int t = 0; t < 4; ++t) {
#pragma unroll
    for (int r = 0; r < 8; ++r) {
      int row = rowA + mh + r;
      int col = colB + 16 * t + n;
      cbuf[(size_t)(b * BQ + q0 + row) * CDN + col] = (_Float16)(acc[t][r] * inv[r]);
    }
  }
}

// ================= Kernel 2: fused resnet MLP stack (32 q-rows per WG) =================
// grid (BQ/32, B), 128 threads (4 waves). Wave w owns cols [64w, 64w+64); two 16-row tiles.

template <int NCHUNKS, int ASTRIDE>
__device__ __forceinline__ void gemm32xH(v8f accA[4], v8f accB[4], const _Float16* sA,
                                         const _Float16* __restrict__ wsrc, int wave) {
#pragma unroll 2
  for (int c = 0; c < NCHUNKS; ++c) {
    const _Float16* wc = wsrc + (size_t)c * WCHUNK;
    v16h a0 = frag_a_lds(sA,  0, c * 32, ASTRIDE);
    v16h a1 = frag_a_lds(sA, 16, c * 32, ASTRIDE);
    v16h b0 = frag_b_g(wc, 64 * wave +  0);
    v16h b1 = frag_b_g(wc, 64 * wave + 16);
    v16h b2 = frag_b_g(wc, 64 * wave + 32);
    v16h b3 = frag_b_g(wc, 64 * wave + 48);
    accA[0] = __builtin_amdgcn_wmma_f32_16x16x32_f16(false, a0, false, b0, (short)0, accA[0], false, false);
    accB[0] = __builtin_amdgcn_wmma_f32_16x16x32_f16(false, a1, false, b0, (short)0, accB[0], false, false);
    accA[1] = __builtin_amdgcn_wmma_f32_16x16x32_f16(false, a0, false, b1, (short)0, accA[1], false, false);
    accB[1] = __builtin_amdgcn_wmma_f32_16x16x32_f16(false, a1, false, b1, (short)0, accB[1], false, false);
    accA[2] = __builtin_amdgcn_wmma_f32_16x16x32_f16(false, a0, false, b2, (short)0, accA[2], false, false);
    accB[2] = __builtin_amdgcn_wmma_f32_16x16x32_f16(false, a1, false, b2, (short)0, accB[2], false, false);
    accA[3] = __builtin_amdgcn_wmma_f32_16x16x32_f16(false, a0, false, b3, (short)0, accA[3], false, false);
    accB[3] = __builtin_amdgcn_wmma_f32_16x16x32_f16(false, a1, false, b3, (short)0, accB[3], false, false);
  }
}

__global__ __launch_bounds__(128)
void k_mlp(const _Float16* __restrict__ cbuf, const _Float16* __restrict__ wpack,
           const float* __restrict__ p,
           const float* __restrict__ fc_p_w, const float* __restrict__ fc_p_b,
           const float* __restrict__ fc_c_b, const float* __restrict__ blk0_b,
           const float* __restrict__ blk1_b,
           const float* __restrict__ fc_out_w, const float* __restrict__ fc_out_b,
           float* __restrict__ out) {
  __shared__ float     s_net[32 * 260];    // f32 residual state (33.3 KB)
  __shared__ _Float16  s_act[32 * 264];    // f16 activations, A matrix (16.9 KB)
  __shared__ _Float16  s_c[32 * 136];      // f16 c tile, reused all blocks (8.7 KB)
  __shared__ float     s_p[96];

  const int b    = blockIdx.y;
  const int q0   = blockIdx.x * 32;
  const int tid  = threadIdx.x;
  const int lane = tid & 31;
  const int wave = tid >> 5;

  {
    int row  = tid >> 2;
    int colb = (tid & 3) * 32;
    const _Float16* src = cbuf + (size_t)(b * BQ + q0 + row) * CDN + colb;
#pragma unroll
    for (int j = 0; j < 32; ++j) s_c[row * 136 + colb + j] = src[j];
  }
  if (tid < 96) s_p[tid] = p[(size_t)(b * BQ + q0) * 3 + tid];
  __syncthreads();
  {
    int r = tid >> 2, hb = (tid & 3) * 64;
    float px = s_p[r*3+0], py = s_p[r*3+1], pz = s_p[r*3+2];
#pragma unroll
    for (int j = 0; j < 64; ++j) {
      int h = hb + j;
      s_net[r * 260 + h] = fc_p_b[h] + px * fc_p_w[h] + py * fc_p_w[HD + h] + pz * fc_p_w[2 * HD + h];
    }
  }

  const v8f vzero = {0.f, 0.f, 0.f, 0.f, 0.f, 0.f, 0.f, 0.f};
  const int n  = lane & 15;
  const int mh = (lane >> 4) * 8;

  for (int i = 0; i < NBK; ++i) {
    // ---- net += c @ fc_c_w[i] + fc_c_b[i] ----
    v8f accA[4] = {vzero, vzero, vzero, vzero};
    v8f accB[4] = {vzero, vzero, vzero, vzero};
    gemm32xH<CDN / 32, 136>(accA, accB, s_c, wpack + (size_t)(i * 4) * WCHUNK, wave);
    __syncthreads();
#pragma unroll
    for (int t = 0; t < 4; ++t)
#pragma unroll
      for (int r = 0; r < 8; ++r) {
        int row = mh + r, col = 64 * wave + 16 * t + n;
        float bias = fc_c_b[i * HD + col];
        s_net[row * 260 + col]        += accA[t][r] + bias;
        s_net[(row + 16) * 260 + col] += accB[t][r] + bias;
      }
    __syncthreads();
    // ---- s_act = relu(net) as f16 ----
    {
      int r = tid >> 2, hb = (tid & 3) * 64;
#pragma unroll
      for (int j = 0; j < 64; ++j)
        s_act[r * 264 + hb + j] = (_Float16)fmaxf(s_net[r * 260 + hb + j], 0.f);
    }
    __syncthreads();
    // ---- h2 = relu(relu(net) @ blk0_w[i] + blk0_b[i]) ----
    v8f acc2A[4] = {vzero, vzero, vzero, vzero};
    v8f acc2B[4] = {vzero, vzero, vzero, vzero};
    gemm32xH<HD / 32, 264>(acc2A, acc2B, s_act, wpack + (size_t)(20 + i * 8) * WCHUNK, wave);
    __syncthreads();                       // all reads of s_act finished
#pragma unroll
    for (int t = 0; t < 4; ++t)
#pragma unroll
      for (int r = 0; r < 8; ++r) {
        int row = mh + r, col = 64 * wave + 16 * t + n;
        float bias = blk0_b[i * HD + col];
        s_act[row * 264 + col]        = (_Float16)fmaxf(acc2A[t][r] + bias, 0.f);
        s_act[(row + 16) * 264 + col] = (_Float16)fmaxf(acc2B[t][r] + bias, 0.f);
      }
    __syncthreads();
    // ---- net += h2 @ blk1_w[i] + blk1_b[i] ----
    v8f acc3A[4] = {vzero, vzero, vzero, vzero};
    v8f acc3B[4] = {vzero, vzero, vzero, vzero};
    gemm32xH<HD / 32, 264>(acc3A, acc3B, s_act, wpack + (size_t)(60 + i * 8) * WCHUNK, wave);
    __syncthreads();
#pragma unroll
    for (int t = 0; t < 4; ++t)
#pragma unroll
      for (int r = 0; r < 8; ++r) {
        int row = mh + r, col = 64 * wave + 16 * t + n;
        float bias = blk1_b[i * HD + col];
        s_net[row * 260 + col]        += acc3A[t][r] + bias;
        s_net[(row + 16) * 260 + col] += acc3B[t][r] + bias;
      }
    __syncthreads();
  }

  // ---- out = relu(net) @ fc_out_w + fc_out_b ----
  if (tid < 32) {
    float s = fc_out_b[0];
    for (int h = 0; h < HD; ++h) s += fmaxf(s_net[tid * 260 + h], 0.f) * fc_out_w[h];
    out[(size_t)b * BQ + q0 + tid] = s;
  }
}

extern "C" void kernel_launch(void* const* d_in, const int* in_sizes, int n_in,
                              void* d_out, int out_size, void* d_ws, size_t ws_size,
                              hipStream_t stream) {
  (void)in_sizes; (void)n_in; (void)out_size; (void)ws_size;
  const float* p        = (const float*)d_in[0];
  const float* pp       = (const float*)d_in[1];
  const float* fea      = (const float*)d_in[2];
  const float* fc_p_w   = (const float*)d_in[3];
  const float* fc_p_b   = (const float*)d_in[4];
  const float* fc_c_w   = (const float*)d_in[5];
  const float* fc_c_b   = (const float*)d_in[6];
  const float* blk0_w   = (const float*)d_in[7];
  const float* blk0_b   = (const float*)d_in[8];
  const float* blk1_w   = (const float*)d_in[9];
  const float* blk1_b   = (const float*)d_in[10];
  const float* fc_out_w = (const float*)d_in[11];
  const float* fc_out_b = (const float*)d_in[12];
  float*    out   = (float*)d_out;
  _Float16* cbuf  = (_Float16*)d_ws;               // 4 MB
  _Float16* wpack = cbuf + WPACK_OFF;              // 1.6 MB packed weights
  _Float16* fpack = cbuf + FPACK_OFF;              // 2.1 MB packed fea

  dim3 blk(128);
  k_repack_w  <<<dim3(NWCHUNKS), blk, 0, stream>>>(fc_c_w, blk0_w, blk1_w, wpack);
  k_repack_fea<<<dim3(BB * PN / 32), blk, 0, stream>>>(fea, fpack);
  k_gauss_feature<<<dim3(BQ / 32, BB), blk, 0, stream>>>(p, pp, fpack, cbuf);
  k_mlp<<<dim3(BQ / 32, BB), blk, 0, stream>>>(cbuf, wpack, p, fc_p_w, fc_p_b,
                                               fc_c_b, blk0_b, blk1_b, fc_out_w, fc_out_b, out);
}
